// pbCosineSim_23201413333423
// MI455X (gfx1250) — matrix operand order repaired and checked by reading, not compiled, not run on hardware
//
#include <hip/hip_runtime.h>

// Batch pairwise cosine similarity: x (8,2048,512) f32 -> sim (8,2048,2048) f32
// sim[b,i,j] = max( dot(x_bi, x_bj) / (|x_bi| |x_bj|), 1e-6 )
//
// Per-batch GEMM X*X^T with V_WMMA_F32_16X16X32_BF16 using a precomputed
// fp32->bf16 hi/lo split (hi*hi + hi*lo + lo*hi, f32 accumulate) for
// near-fp32 accuracy at ~8/3 the f32-WMMA rate. The block-shared A panel is
// staged into LDS via double-buffered GLOBAL_LOAD_ASYNC_TO_LDS_B128
// (ASYNCcnt) to remove the 8x redundant A reads; per-wave-unique B rows load
// straight from L2 (entire bf16 operand set fits in the 192MB L2).

typedef __attribute__((ext_vector_type(16))) __bf16 v16bf;
typedef __attribute__((ext_vector_type(8)))  __bf16 v8bf;
typedef __attribute__((ext_vector_type(4)))  __bf16 v4bf;
typedef __attribute__((ext_vector_type(8)))  float  v8f;

#define BATCH 8
#define SEQ   2048
#define DIM   512
#define EPS   1e-6f

#define AROW_STRIDE 40   // __bf16 elements per LDS A-panel row (80 B, padded)

static __device__ __forceinline__ v16bf cat16(v8bf a, v8bf b) {
    return __builtin_shufflevector(a, b, 0, 1, 2, 3, 4, 5, 6, 7,
                                         8, 9, 10, 11, 12, 13, 14, 15);
}

// ------------- prep: one wave per row; bf16 hi/lo split + 1/||row|| -------------
__global__ __launch_bounds__(256) void prep_k(const float* __restrict__ x,
                                              __bf16* __restrict__ hi,
                                              __bf16* __restrict__ lo,
                                              float* __restrict__ rnorm) {
    const int row  = (blockIdx.x * 256 + threadIdx.x) >> 5;   // 0..16383
    const int lane = threadIdx.x & 31;
    const float*  src = x  + (size_t)row * DIM;
    __bf16*       hr  = hi + (size_t)row * DIM;
    __bf16*       lr  = lo + (size_t)row * DIM;

    float s = 0.0f;
#pragma unroll
    for (int j = 0; j < 4; ++j) {
        const int idx = lane * 4 + j * 128;
        const float4 v = *(const float4*)(src + idx);
        s += v.x * v.x + v.y * v.y + v.z * v.z + v.w * v.w;

        v4bf h, l;
        const float f[4] = {v.x, v.y, v.z, v.w};
#pragma unroll
        for (int e = 0; e < 4; ++e) {
            const __bf16 hb = (__bf16)f[e];
            h[e] = hb;
            l[e] = (__bf16)(f[e] - (float)hb);
        }
        *(v4bf*)(hr + idx) = h;
        *(v4bf*)(lr + idx) = l;
    }
#pragma unroll
    for (int off = 16; off > 0; off >>= 1)
        s += __shfl_xor(s, off, 32);
    if (lane == 0) rnorm[row] = 1.0f / sqrtf(s);
}

// ------------- fallback: norms only (when scratch can't hold hi/lo) -------------
__global__ __launch_bounds__(256) void rownorm_k(const float* __restrict__ x,
                                                 float* __restrict__ rnorm) {
    const int row  = (blockIdx.x * 256 + threadIdx.x) >> 5;
    const int lane = threadIdx.x & 31;
    const float* src = x + (size_t)row * DIM;
    float s = 0.0f;
#pragma unroll
    for (int j = 0; j < 4; ++j) {
        const float4 v = *(const float4*)(src + (lane + j * 32) * 4);
        s += v.x * v.x + v.y * v.y + v.z * v.z + v.w * v.w;
    }
#pragma unroll
    for (int off = 16; off > 0; off >>= 1)
        s += __shfl_xor(s, off, 32);
    if (lane == 0) rnorm[row] = 1.0f / sqrtf(s);
}

// shared epilogue: f32 C/D layout -> VGPR r, lane half h: M = 8h + r, N = lane%16
static __device__ __forceinline__ void epilogue(const v8f acc[4][4],
                                                const float* __restrict__ rnb,
                                                float* __restrict__ out,
                                                int b, int m0, int n0,
                                                int half, int r16) {
    float rnj[4];
#pragma unroll
    for (int nt = 0; nt < 4; ++nt) rnj[nt] = rnb[n0 + nt * 16 + r16];

#pragma unroll
    for (int mt = 0; mt < 4; ++mt) {
        const int ibase = m0 + mt * 16 + 8 * half;
        const v8f rni = *(const v8f*)(rnb + ibase);
#pragma unroll
        for (int nt = 0; nt < 4; ++nt) {
            const int j = n0 + nt * 16 + r16;
            const float sj = rnj[nt];
#pragma unroll
            for (int r = 0; r < 8; ++r) {
                float v = acc[mt][nt][r] * rni[r] * sj;
                v = fmaxf(v, EPS);
                __builtin_nontemporal_store(
                    v, out + ((size_t)b * SEQ + (ibase + r)) * SEQ + j);
            }
        }
    }
}

// ------ main GEMM: async-LDS A panel + precomputed bf16, one wave per 64x64 tile ------
__global__ __launch_bounds__(256) void cosgemm_pre_k(const __bf16* __restrict__ hi,
                                                     const __bf16* __restrict__ lo,
                                                     const float* __restrict__ rnorm,
                                                     float* __restrict__ out) {
    // A panel double buffer: [buf][hi/lo][64 rows * 40 elems] (80B padded rows)
    __shared__ __align__(16) __bf16 smA[2][2][64 * AROW_STRIDE];

    const int lane = threadIdx.x & 31;
    const int half = lane >> 4;
    const int r16  = lane & 15;

    const int w   = blockIdx.x * 8 + (threadIdx.x >> 5);
    const int tps = SEQ / 64;                    // 32 n-tiles per m-row
    const int b   = w / (tps * tps);             // block-uniform (8 | 1024)
    const int rem = w - b * tps * tps;
    const int m0  = (rem / tps) * 64;            // block-uniform (8 | 32)
    const int n0  = (rem % tps) * 64;            // per-wave

    const __bf16* Hb  = hi + (size_t)b * SEQ * DIM;
    const __bf16* Lb  = lo + (size_t)b * SEQ * DIM;
    const float*  rnb = rnorm + b * SEQ;

    // async A-panel staging: thread t covers row t/4, k-segment (t%4)*8 (16 B)
    const int arow = threadIdx.x >> 2;
    const int aseg = threadIdx.x & 3;
    const size_t gAoff = (size_t)(m0 + arow) * DIM + aseg * 8;
    const unsigned ldsAoff = (unsigned)(arow * AROW_STRIDE + aseg * 8) * sizeof(__bf16);

    auto issueA = [&](int buf, int k0) {
        const __bf16* gH = Hb + gAoff + k0;
        const __bf16* gL = Lb + gAoff + k0;
        const unsigned dH = (unsigned)(uintptr_t)(&smA[buf][0][0]) + ldsAoff;
        const unsigned dL = (unsigned)(uintptr_t)(&smA[buf][1][0]) + ldsAoff;
        asm volatile("global_load_async_to_lds_b128 %0, %1, off"
                     :: "v"(dH), "v"(gH) : "memory");
        asm volatile("global_load_async_to_lds_b128 %0, %1, off"
                     :: "v"(dL), "v"(gL) : "memory");
    };

    v8f acc[4][4] = {};

    issueA(0, 0);                                   // prologue prefetch
    for (int step = 0; step < DIM / 32; ++step) {
        const int cur = step & 1;
        const int k0  = step * 32;

        if (step + 1 < DIM / 32) {
            issueA(cur ^ 1, k0 + 32);
            asm volatile("s_wait_asynccnt 0x2" ::: "memory");  // cur buffer landed
        } else {
            asm volatile("s_wait_asynccnt 0x0" ::: "memory");
        }
        __syncthreads();                            // cur panel visible to all waves

        // A frags from LDS (16-bit 16x32 layout): runs [8h..+7], [16+8h..+7]
        v16bf hiA[4], loA[4];
        const __bf16* As = &smA[cur][0][0];
        const __bf16* Al = &smA[cur][1][0];
#pragma unroll
        for (int mt = 0; mt < 4; ++mt) {
            const int rb = (mt * 16 + r16) * AROW_STRIDE + 8 * half;
            hiA[mt] = cat16(*(const v8bf*)(As + rb), *(const v8bf*)(As + rb + 16));
            loA[mt] = cat16(*(const v8bf*)(Al + rb), *(const v8bf*)(Al + rb + 16));
        }

        // B frags straight from global (L2-resident), runs [8h..+7], [16+8h..+7]
#pragma unroll
        for (int nt = 0; nt < 4; ++nt) {
            const size_t off = (size_t)(n0 + nt * 16 + r16) * DIM + k0 + 8 * half;
            const v16bf hiB = cat16(*(const v8bf*)(Hb + off), *(const v8bf*)(Hb + off + 16));
            const v16bf loB = cat16(*(const v8bf*)(Lb + off), *(const v8bf*)(Lb + off + 16));
#pragma unroll
            for (int mt = 0; mt < 4; ++mt) {
                acc[mt][nt] = __builtin_amdgcn_wmma_f32_16x16x32_bf16(
                    false, hiA[mt], false, hiB, (short)0, acc[mt][nt], false, false);
                acc[mt][nt] = __builtin_amdgcn_wmma_f32_16x16x32_bf16(
                    false, hiA[mt], false, loB, (short)0, acc[mt][nt], false, false);
                acc[mt][nt] = __builtin_amdgcn_wmma_f32_16x16x32_bf16(
                    false, loA[mt], false, hiB, (short)0, acc[mt][nt], false, false);
            }
        }

        __syncthreads();      // everyone done reading cur before it is overwritten
    }

    epilogue(acc, rnb, out, b, m0, n0, half, r16);
}

// ------------- fallback GEMM: fp32 loads, in-loop hi/lo conversion -------------
__global__ __launch_bounds__(256) void cosgemm_fly_k(const float* __restrict__ x,
                                                     const float* __restrict__ rnorm,
                                                     float* __restrict__ out) {
    const int lane = threadIdx.x & 31;
    const int half = lane >> 4;
    const int r16  = lane & 15;

    const int w   = blockIdx.x * 8 + (threadIdx.x >> 5);
    const int tps = SEQ / 64;
    const int b   = w / (tps * tps);
    const int rem = w - b * tps * tps;
    const int m0  = (rem / tps) * 64;
    const int n0  = (rem % tps) * 64;

    const float* Xb  = x + (size_t)b * SEQ * DIM;
    const float* rnb = rnorm + b * SEQ;

    v8f acc[4][4] = {};

    for (int k0 = 0; k0 < DIM; k0 += 32) {
        v16bf hiA[4], loA[4];
#pragma unroll
        for (int mt = 0; mt < 4; ++mt) {
            const float* p = Xb + (size_t)(m0 + mt * 16 + r16) * DIM + k0 + 8 * half;
#pragma unroll
            for (int e = 0; e < 8; ++e) {
                const float f0 = p[e];
                const float f1 = p[e + 16];
                const __bf16 h0 = (__bf16)f0;
                const __bf16 h1 = (__bf16)f1;
                hiA[mt][e]     = h0;
                loA[mt][e]     = (__bf16)(f0 - (float)h0);
                hiA[mt][e + 8] = h1;
                loA[mt][e + 8] = (__bf16)(f1 - (float)h1);
            }
        }
#pragma unroll
        for (int nt = 0; nt < 4; ++nt) {
            const float* q = Xb + (size_t)(n0 + nt * 16 + r16) * DIM + k0 + 8 * half;
            v16bf hiB, loB;
#pragma unroll
            for (int e = 0; e < 16; ++e) {
                const float f = q[(e & 7) + ((e >> 3) << 4)];
                const __bf16 h = (__bf16)f;
                hiB[e] = h;
                loB[e] = (__bf16)(f - (float)h);
            }
#pragma unroll
            for (int mt = 0; mt < 4; ++mt) {
                acc[mt][nt] = __builtin_amdgcn_wmma_f32_16x16x32_bf16(
                    false, hiA[mt], false, hiB, (short)0, acc[mt][nt], false, false);
                acc[mt][nt] = __builtin_amdgcn_wmma_f32_16x16x32_bf16(
                    false, hiA[mt], false, loB, (short)0, acc[mt][nt], false, false);
                acc[mt][nt] = __builtin_amdgcn_wmma_f32_16x16x32_bf16(
                    false, loA[mt], false, hiB, (short)0, acc[mt][nt], false, false);
            }
        }
    }

    epilogue(acc, rnb, out, b, m0, n0, half, r16);
}

extern "C" void kernel_launch(void* const* d_in, const int* in_sizes, int n_in,
                              void* d_out, int out_size, void* d_ws, size_t ws_size,
                              hipStream_t stream) {
    const float* x   = (const float*)d_in[0];
    float*       out = (float*)d_out;

    const size_t nElem   = (size_t)BATCH * SEQ * DIM;
    const size_t rnBytes = (size_t)BATCH * SEQ * sizeof(float);
    const size_t need    = rnBytes + 2 * nElem * sizeof(__bf16);   // ~33.6 MB

    float* rn = (float*)d_ws;
    const dim3 blk(256);
    const dim3 normGrid((BATCH * SEQ) / 8);
    const dim3 gemmGrid((BATCH * 32 * 32) / 8);

    if (ws_size >= need) {
        __bf16* hi = (__bf16*)((char*)d_ws + rnBytes);
        __bf16* lo = hi + nElem;
        prep_k<<<normGrid, blk, 0, stream>>>(x, hi, lo, rn);
        cosgemm_pre_k<<<gemmGrid, blk, 0, stream>>>(hi, lo, rn, out);
    } else {
        rownorm_k<<<normGrid, blk, 0, stream>>>(x, rn);
        cosgemm_fly_k<<<gemmGrid, blk, 0, stream>>>(x, rn, out);
    }
}
